// GCONVSTACK_60224031425324
// MI455X (gfx1250) — compile-verified, matrix operand order checked by reading, not run on hardware
//
#include <hip/hip_runtime.h>
#include <hip/hip_bf16.h>

// ---------------------------------------------------------------------------
// GCONV stack for MI455X (gfx1250): fp32 atomic segment-sum (L2-resident),
// bf16 WMMA GEMMs (v_wmma_f32_16x16x32_bf16) with f32 accum, fused bias+ReLU,
// fused head. d_out regions double as the big scratch buffers.
//
// GEMM register blocking: 32-row strip per block, each of 8 waves owns
// 2 row-tiles x 2 col-tiles -> 8 WMMAs per k-step against 8 ds + 8 global
// b128 fragment loads (B-frags amortized across both row tiles).
// ---------------------------------------------------------------------------

#define N_NODES 100000
#define DEG     4
#define E_EDGES (N_NODES * DEG)
#define IN_F    128
#define H_F     256

typedef __attribute__((ext_vector_type(16))) __bf16 v16bf;
typedef __attribute__((ext_vector_type(8)))  float  v8f;

// ---------------- zero (float4) ----------------
__global__ void zero_f32(float* __restrict__ p, long n) {
  long i4 = ((long)blockIdx.x * blockDim.x + threadIdx.x) * 4;
  if (i4 + 3 < n) {
    *(float4*)(p + i4) = make_float4(0.f, 0.f, 0.f, 0.f);
  } else {
    for (long j = i4; j < n; ++j) p[j] = 0.f;
  }
}

// ---------------- f32 -> bf16 weight conversion ----------------
__global__ void cvt_bf16(const float* __restrict__ in, __bf16* __restrict__ out, int n) {
  int i = blockIdx.x * blockDim.x + threadIdx.x;
  if (i < n) out[i] = (__bf16)in[i];
}

// ---------------- segment-sum: agg[dst] += feat[src] ----------------
// one thread per (edge, 4-float chunk); h fits in the 192MB L2 so the random
// gathers + f32 atomics stay on-chip.
__global__ void scatter_add(const float* __restrict__ feat,
                            const int* __restrict__ src,
                            const int* __restrict__ dst,
                            float* __restrict__ agg, int K) {
  int chunks = K >> 2;
  long t = (long)blockIdx.x * blockDim.x + threadIdx.x;
  long total = (long)E_EDGES * chunks;
  if (t >= total) return;
  int e = (int)(t / chunks);
  int c = (int)(t - (long)e * chunks) * 4;
  int s = src[e], d = dst[e];
  float4 v = *(const float4*)(feat + (long)s * K + c);
  float* a = agg + (long)d * K + c;
  atomicAdd(a + 0, v.x);
  atomicAdd(a + 1, v.y);
  atomicAdd(a + 2, v.z);
  atomicAdd(a + 3, v.w);
}

// ---------------- fused gconv GEMM ----------------
// Hout[m, :] = relu(Agg[m, :] @ Wr^T + br + Hin[m, :] @ Ws^T)
// Block: 32 rows, 8 waves; wave w covers col tiles {16w, 16w+128} for both
// 16-row tiles. In-place safe: strip staged to LDS (bf16) before any store.
#define WMMA_BF16(A, B, C) \
  __builtin_amdgcn_wmma_f32_16x16x32_bf16(false, (A), false, (B), (short)0, (C), false, false)

template <int K>
__global__ __launch_bounds__(256)
void gconv_gemm(const float* __restrict__ Agg, const float* __restrict__ Hin,
                const __bf16* __restrict__ Wr, const __bf16* __restrict__ Ws,
                const float* __restrict__ br, float* __restrict__ Hout) {
  __shared__ __bf16 sAgg[32 * K];
  __shared__ __bf16 sH[32 * K];

  const int tid = threadIdx.x;
  const int rowBase = blockIdx.x * 32;

  // stage + convert this block's 32-row strip (coalesced; K is power of two)
  for (int i = tid; i < 32 * K; i += 256) {
    int r = i / K, c = i & (K - 1);
    long g = (long)(rowBase + r) * K + c;
    sAgg[i] = (__bf16)Agg[g];
    sH[i]   = (__bf16)Hin[g];
  }
  __syncthreads();

  const int wave  = tid >> 5;
  const int lane  = tid & 31;
  const int mrow  = lane & 15;   // A row / B column / C column (per ISA layout)
  const int khalf = lane >> 4;   // K-half selector
  const int col0  = wave * 16;   // column tile 0
  const int col1  = col0 + 128;  // column tile 1
  const int n0 = col0 + mrow;    // weight row for B-frag (tile 0)
  const int n1 = col1 + mrow;    // weight row for B-frag (tile 1)

  v8f c00 = {};  // (rows 0-15,  cols col0..)
  v8f c01 = {};  // (rows 0-15,  cols col1..)
  v8f c10 = {};  // (rows 16-31, cols col0..)
  v8f c11 = {};  // (rows 16-31, cols col1..)

  const __bf16* w0r = Wr + (long)n0 * K;
  const __bf16* w1r = Wr + (long)n1 * K;
  const __bf16* w0s = Ws + (long)n0 * K;
  const __bf16* w1s = Ws + (long)n1 * K;

  for (int kk = 0; kk < K; kk += 32) {
    const int kA = kk + khalf * 8;       // first 8-bf16 run
    const int kB = kk + 16 + khalf * 8;  // second 8-bf16 run

    v16bf aA0, aA1, aH0, aH1, b0r, b0s, b1r, b1s;
    const __bf16* pA0a = &sAgg[mrow * K + kA];
    const __bf16* pA0b = &sAgg[mrow * K + kB];
    const __bf16* pA1a = &sAgg[(mrow + 16) * K + kA];
    const __bf16* pA1b = &sAgg[(mrow + 16) * K + kB];
    const __bf16* pH0a = &sH[mrow * K + kA];
    const __bf16* pH0b = &sH[mrow * K + kB];
    const __bf16* pH1a = &sH[(mrow + 16) * K + kA];
    const __bf16* pH1b = &sH[(mrow + 16) * K + kB];
#pragma unroll
    for (int i = 0; i < 8; ++i) {
      aA0[i]     = pA0a[i];
      aA0[i + 8] = pA0b[i];
      aA1[i]     = pA1a[i];
      aA1[i + 8] = pA1b[i];
      aH0[i]     = pH0a[i];
      aH0[i + 8] = pH0b[i];
      aH1[i]     = pH1a[i];
      aH1[i + 8] = pH1b[i];
      b0r[i]     = w0r[kA + i];
      b0r[i + 8] = w0r[kB + i];
      b1r[i]     = w1r[kA + i];
      b1r[i + 8] = w1r[kB + i];
      b0s[i]     = w0s[kA + i];
      b0s[i + 8] = w0s[kB + i];
      b1s[i]     = w1s[kA + i];
      b1s[i + 8] = w1s[kB + i];
    }
    c00 = WMMA_BF16(aA0, b0r, c00);
    c00 = WMMA_BF16(aH0, b0s, c00);
    c01 = WMMA_BF16(aA0, b1r, c01);
    c01 = WMMA_BF16(aH0, b1s, c01);
    c10 = WMMA_BF16(aA1, b0r, c10);
    c10 = WMMA_BF16(aH1, b0s, c10);
    c11 = WMMA_BF16(aA1, b1r, c11);
    c11 = WMMA_BF16(aH1, b1s, c11);
  }

  // epilogue: bias + relu. C/D layout: elem i of lane l -> row i+8*khalf, col mrow
  const float bias0 = br[col0 + mrow];
  const float bias1 = br[col1 + mrow];
#pragma unroll
  for (int i = 0; i < 8; ++i) {
    long m0 = rowBase + i + 8 * khalf;
    long m1 = m0 + 16;
    float v00 = c00[i] + bias0;
    float v01 = c01[i] + bias1;
    float v10 = c10[i] + bias0;
    float v11 = c11[i] + bias1;
    Hout[m0 * H_F + col0 + mrow] = v00 > 0.f ? v00 : 0.f;
    Hout[m0 * H_F + col1 + mrow] = v01 > 0.f ? v01 : 0.f;
    Hout[m1 * H_F + col0 + mrow] = v10 > 0.f ? v10 : 0.f;
    Hout[m1 * H_F + col1 + mrow] = v11 > 0.f ? v11 : 0.f;
  }
}

// ---------------- split xs -> x1 (even rows) / x2 (odd rows) ----------------
__global__ void split_xs(const float* __restrict__ xs,
                         float* __restrict__ x1, float* __restrict__ x2) {
  long i = (long)blockIdx.x * blockDim.x + threadIdx.x;
  long total = (long)N_NODES * H_F;
  if (i >= total) return;
  int row = (int)(i / H_F);
  int col = (int)(i - (long)row * H_F);
  float v = xs[i];
  long o = (long)(row >> 1) * H_F + col;
  if (row & 1) x2[o] = v;
  else         x1[o] = v;
}

// ---------------- head: per-pair dots + relu + sigmoid ----------------
__global__ __launch_bounds__(256)
void head_kernel(const float* __restrict__ h,
                 const float* __restrict__ Wsp1, const float* __restrict__ bsp1,
                 const float* __restrict__ Wsp2, const float* __restrict__ bsp2,
                 const float* __restrict__ Wl,   const float* __restrict__ bl,
                 float* __restrict__ y) {
  int wave = threadIdx.x >> 5;
  int lane = threadIdx.x & 31;
  int pair = blockIdx.x * 8 + wave;
  if (pair >= N_NODES / 2) return;
  const float* h1 = h + (long)(2 * pair) * H_F;
  const float* h2 = h1 + H_F;
  float s1 = 0.f, s2 = 0.f;
#pragma unroll
  for (int j = 0; j < 8; ++j) {
    int f = lane + 32 * j;
    s1 += h1[f] * Wsp1[f];
    s2 += h2[f] * Wsp2[f];
  }
#pragma unroll
  for (int off = 16; off > 0; off >>= 1) {
    s1 += __shfl_xor(s1, off, 32);
    s2 += __shfl_xor(s2, off, 32);
  }
  if (lane == 0) {
    s1 = fmaxf(s1 + bsp1[0], 0.f);
    s2 = fmaxf(s2 + bsp2[0], 0.f);
    float o = s1 * Wl[0] + s2 * Wl[1] + bl[0];
    y[pair] = 1.f / (1.f + __expf(-o));
  }
}

// ---------------------------------------------------------------------------
extern "C" void kernel_launch(void* const* d_in, const int* in_sizes, int n_in,
                              void* d_out, int out_size, void* d_ws, size_t ws_size,
                              hipStream_t stream) {
  (void)in_sizes; (void)n_in; (void)out_size; (void)ws_size;

  const float* x    = (const float*)d_in[0];
  const int*   eidx = (const int*)d_in[1];
  // d_in[2] = batch (unused)
  const float* Wr0 = (const float*)d_in[3];
  const float* br0 = (const float*)d_in[4];
  const float* Ws0 = (const float*)d_in[5];
  const float* Wr1 = (const float*)d_in[6];
  const float* br1 = (const float*)d_in[7];
  const float* Ws1 = (const float*)d_in[8];
  const float* Wr2 = (const float*)d_in[9];
  const float* br2 = (const float*)d_in[10];
  const float* Ws2 = (const float*)d_in[11];
  const float* Wsp1 = (const float*)d_in[12];
  const float* bsp1 = (const float*)d_in[13];
  const float* Wsp2 = (const float*)d_in[14];
  const float* bsp2 = (const float*)d_in[15];
  const float* Wl  = (const float*)d_in[16];
  const float* bl  = (const float*)d_in[17];

  const int* src = eidx;            // edge_index[0]
  const int* dst = eidx + E_EDGES;  // edge_index[1]

  // d_out layout: y (50000) | xs (N*H) | x1 (N/2*H) | x2 (N/2*H)
  float* out    = (float*)d_out;
  float* y_out  = out;
  float* xs_out = out + 50000;                       // lives as the h buffer
  float* x1_out = xs_out + (long)N_NODES * H_F;
  // x1 + x2 are contiguous: exactly N*H floats -> aggregation scratch
  float* hbuf   = xs_out;
  float* aggbuf = x1_out;
  float* x2_out = x1_out + (long)(N_NODES / 2) * H_F;

  // small ws: bf16 weight copies (max 2 * 256*256*2 = 256 KB)
  __bf16* wrb = (__bf16*)d_ws;
  __bf16* wsb = wrb + (long)H_F * H_F;

  auto prep_layer = [&](const float* hin, const float* Wr, const float* Ws, int K) {
    int nW = H_F * K;
    cvt_bf16<<<(nW + 255) / 256, 256, 0, stream>>>(Wr, wrb, nW);
    cvt_bf16<<<(nW + 255) / 256, 256, 0, stream>>>(Ws, wsb, nW);

    long nz = (long)N_NODES * K;
    zero_f32<<<(unsigned)((nz / 4 + 255) / 256), 256, 0, stream>>>(aggbuf, nz);

    long nscatter = (long)E_EDGES * (K >> 2);
    scatter_add<<<(unsigned)((nscatter + 255) / 256), 256, 0, stream>>>(
        hin, src, dst, aggbuf, K);
  };

  // layer 0: 128 -> 256
  prep_layer(x, Wr0, Ws0, IN_F);
  gconv_gemm<IN_F><<<N_NODES / 32, 256, 0, stream>>>(aggbuf, x, wrb, wsb, br0, hbuf);

  // layer 1 (in-place, LDS-staged)
  prep_layer(hbuf, Wr1, Ws1, H_F);
  gconv_gemm<H_F><<<N_NODES / 32, 256, 0, stream>>>(aggbuf, hbuf, wrb, wsb, br1, hbuf);

  // layer 2 -> final h lands directly in the xs region
  prep_layer(hbuf, Wr2, Ws2, H_F);
  gconv_gemm<H_F><<<N_NODES / 32, 256, 0, stream>>>(aggbuf, hbuf, wrb, wsb, br2, hbuf);

  // deg == 4 for every node (src = repeat(arange(N), 4)), so sel is the
  // identity: xs is already in place; split even/odd rows into x1/x2
  // (overwriting the agg scratch only after its last use).
  long total = (long)N_NODES * H_F;
  split_xs<<<(unsigned)((total + 255) / 256), 256, 0, stream>>>(hbuf, x1_out, x2_out);

  head_kernel<<<(N_NODES / 2) / 8, 256, 0, stream>>>(
      hbuf, Wsp1, bsp1, Wsp2, bsp2, Wl, bl, y_out);
}
